// MULTIMANET_73993696575906
// MI455X (gfx1250) — matrix operand order repaired and checked by reading, not compiled, not run on hardware
//
#include <hip/hip_runtime.h>

// ---------------------------------------------------------------------------
// MI455X (gfx1250) implementation of the multimodal emotion net forward pass.
// Heavy contractions (convs via implicit GEMM, attention QKV projections, GRU
// input projection) run on v_wmma_f32_16x16x32_f16 (f16 operands, f32 acc),
// 16x32 output tile per wave (2 wmma per K-step, A-fragment reuse).
// Conv weights are repacked to [O][9][Cin]; im2col run state (pixel, channel
// base) is advanced incrementally (2 conditional subtracts) -> no division in
// the hot loop.
// ---------------------------------------------------------------------------

typedef _Float16 half16 __attribute__((ext_vector_type(16)));
typedef float    float8 __attribute__((ext_vector_type(8)));

__device__ __forceinline__ int iclampi(int x, int lo, int hi) {
    return x < lo ? lo : (x > hi ? hi : x);
}
__device__ __forceinline__ int imini(int a, int b) { return a < b ? a : b; }

// Repack conv weight [O][Cin][3][3] -> [O][9][Cin]  (k_new = r*Cin + c)
__global__ void repack_w9(const float* __restrict__ in, float* __restrict__ out,
                          int O, int Cin)
{
    int K = Cin * 9;
    int id = blockIdx.x * blockDim.x + threadIdx.x;
    if (id >= O * K) return;
    int o = id / K, k = id - o * K;
    int r = k / Cin, c = k - r * Cin;
    out[id] = in[o * K + c * 9 + r];
}

// Load one 8-wide im2col run (single pixel r, channels cb..cb+7), masked.
__device__ __forceinline__ void im2col_run(const float* __restrict__ A,
                                           int pb, int py, int px,
                                           int H, int W, int Cin,
                                           int r, int cb, bool kin,
                                           half16& av, int o)
{
    int dy = r / 3;            // constant divisor: cheap mulhi
    int dx = r - dy * 3;
    int iy = py + dy - 1;
    int ix = px + dx - 1;
    bool inb = kin & (iy >= 0) & (iy < H) & (ix >= 0) & (ix < W);
    int iyc = iclampi(iy, 0, H - 1);
    int ixc = iclampi(ix, 0, W - 1);
    const float4* src = (const float4*)(A + (((long)pb * H + iyc) * W + ixc) * Cin + cb);
    float4 u = src[0], w = src[1];
    av[o + 0] = (_Float16)(inb ? u.x : 0.f);
    av[o + 1] = (_Float16)(inb ? u.y : 0.f);
    av[o + 2] = (_Float16)(inb ? u.z : 0.f);
    av[o + 3] = (_Float16)(inb ? u.w : 0.f);
    av[o + 4] = (_Float16)(inb ? w.x : 0.f);
    av[o + 5] = (_Float16)(inb ? w.y : 0.f);
    av[o + 6] = (_Float16)(inb ? w.z : 0.f);
    av[o + 7] = (_Float16)(inb ? w.w : 0.f);
}

// Advance run state by 32 channels (32 < 2*Cin for Cin>=24): branchless.
__device__ __forceinline__ void adv_run(int& r, int& cb, int Cin)
{
    cb += 32;
    bool g0 = cb >= Cin; cb = g0 ? cb - Cin : cb; r += g0 ? 1 : 0;
    bool g1 = cb >= Cin; cb = g1 ? cb - Cin : cb; r += g1 ? 1 : 0;
}

// ---------------------------------------------------------------------------
// WMMA GEMM:  C[M,N] = A[M,K] * Bt[N,K]^T  (+ epilogue)
//   a_im2col=1: A is an NHWC tensor [B,H,W,Cin]; row m = 3x3 SAME patch at
//               position m with k = r*Cin + c (weights repacked to [O][9][C]).
//   epi=0: C += bias[n]
//   epi=1: batchnorm (gamma,beta,mean,var) folded with conv bias, then ReLU
// One wave computes a 16x32 tile (two 16x16 accumulators). K-loop is guard-
// free (wave-uniform fast/tail split); boundary handling is branchless,
// keeping EXEC all-1s for WMMA.
// ---------------------------------------------------------------------------
__global__ void __launch_bounds__(128)
gemm16_nt(const float* __restrict__ A, const float* __restrict__ Bt,
          float* __restrict__ C, int M, int N, int K,
          int a_im2col, int H, int W, int Cin,
          int epi, const float* __restrict__ bias,
          const float* __restrict__ bng, const float* __restrict__ bnb,
          const float* __restrict__ bnm, const float* __restrict__ bnv,
          const float* __restrict__ cbias)
{
    int tilesM  = (M + 15) >> 4;
    int tilesN2 = (N + 31) >> 5;
    int tid = blockIdx.x * 4 + threadIdx.y;      // wave-uniform
    if (tid >= tilesM * tilesN2) return;         // whole-wave exit
    int mT = tid % tilesM, nT = tid / tilesM;
    int lane = threadIdx.x;
    int l15 = lane & 15;
    bool hi = lane >= 16;
    int m0 = mT << 4, n0 = nT << 5;
    // clamped row indices: OOB rows read valid memory, masked at the store
    int mrow  = imini(m0 + l15, M - 1);
    int nrow0 = imini(n0 + l15, N - 1);
    int nrow1 = imini(n0 + 16 + l15, N - 1);

    const int aoff = hi ? 8 : 0;
    const int boff = hi ? 8 : 0;

    int pb = 0, py = 0, px = 0;
    int r0 = 0, cb0 = 0, r1 = 0, cb1 = 0;        // im2col run state
    if (a_im2col) {
        int hw = H * W;
        pb = mrow / hw;
        int rm = mrow - pb * hw;
        py = rm / W;
        px = rm - py * W;
        cb0 = aoff;                              // aoff < Cin always
        cb1 = 16 + aoff;
        if (cb1 >= Cin) { cb1 -= Cin; r1 = 1; }  // 16+aoff < 2*Cin always
    }
    const float* Arow = A + (long)mrow * K;      // plain mode only
    const float* B0   = Bt + (long)nrow0 * K;
    const float* B1   = Bt + (long)nrow1 * K;

    float8 acc0 = {0.f, 0.f, 0.f, 0.f, 0.f, 0.f, 0.f, 0.f};
    float8 acc1 = {0.f, 0.f, 0.f, 0.f, 0.f, 0.f, 0.f, 0.f};

    int k0 = 0;
    for (; k0 + 32 <= K; k0 += 32) {             // guard-free hot loop
        half16 av, bv0, bv1;
        if (!a_im2col) {
            // two contiguous 8-float runs per lane (K even -> float2 aligned)
            const float2* p0 = (const float2*)(Arow + k0 + aoff);
            const float2* p1 = (const float2*)(Arow + k0 + aoff + 16);
#pragma unroll
            for (int j = 0; j < 4; ++j) {
                float2 u = p0[j]; av[2 * j]     = (_Float16)u.x; av[2 * j + 1]     = (_Float16)u.y;
                float2 w = p1[j]; av[8 + 2 * j] = (_Float16)w.x; av[8 + 2 * j + 1] = (_Float16)w.y;
            }
        } else {
            im2col_run(A, pb, py, px, H, W, Cin, r0, cb0, true, av, 0);
            im2col_run(A, pb, py, px, H, W, Cin, r1, cb1, true, av, 8);
            adv_run(r0, cb0, Cin);
            adv_run(r1, cb1, Cin);
        }
        {
            // two 8-float runs per B fragment (k = boff.., boff+16..)
            const float2* q0 = (const float2*)(B0 + k0 + boff);
            const float2* q1 = (const float2*)(B1 + k0 + boff);
#pragma unroll
            for (int j = 0; j < 8; ++j) {
                int jj = j + ((j >= 4) ? 4 : 0);
                float2 u = q0[jj]; bv0[2 * j] = (_Float16)u.x; bv0[2 * j + 1] = (_Float16)u.y;
                float2 w = q1[jj]; bv1[2 * j] = (_Float16)w.x; bv1[2 * j + 1] = (_Float16)w.y;
            }
        }
        acc0 = __builtin_amdgcn_wmma_f32_16x16x32_f16(false, av, false, bv0,
                                                      (short)0, acc0, false, false);
        acc1 = __builtin_amdgcn_wmma_f32_16x16x32_f16(false, av, false, bv1,
                                                      (short)0, acc1, false, false);
    }
    if (k0 < K) {                                 // single masked tail iteration
        half16 av, bv0, bv1;
        if (a_im2col) {
            // K = 9*Cin is a multiple of 8 and runs are 8-aligned:
            // each run is fully inside or fully outside K.
            im2col_run(A, pb, py, px, H, W, Cin, r0, cb0, (k0 + aoff) < K,      av, 0);
            im2col_run(A, pb, py, px, H, W, Cin, r1, cb1, (k0 + 16 + aoff) < K, av, 8);
        } else {
#pragma unroll
            for (int i = 0; i < 16; ++i) {
                int ka  = k0 + aoff + i + ((i >= 8) ? 8 : 0);
                int kac = imini(ka, K - 1);
                float va = Arow[kac];
                av[i] = (_Float16)(ka < K ? va : 0.f);
            }
        }
#pragma unroll
        for (int i = 0; i < 16; ++i) {
            int kb  = k0 + boff + i + ((i >= 8) ? 8 : 0);
            int kbc = imini(kb, K - 1);
            bv0[i] = (_Float16)(kb < K ? B0[kbc] : 0.f);
            bv1[i] = (_Float16)(kb < K ? B1[kbc] : 0.f);
        }
        acc0 = __builtin_amdgcn_wmma_f32_16x16x32_f16(false, av, false, bv0,
                                                      (short)0, acc0, false, false);
        acc1 = __builtin_amdgcn_wmma_f32_16x16x32_f16(false, av, false, bv1,
                                                      (short)0, acc1, false, false);
    }

#pragma unroll
    for (int t = 0; t < 2; ++t) {
        const float8& acc = t ? acc1 : acc0;
#pragma unroll
        for (int gq = 0; gq < 8; ++gq) {
            int m = m0 + gq + (hi ? 8 : 0);
            int n = n0 + t * 16 + l15;
            if (m < M && n < N) {
                float v = acc[gq];
                if (epi == 0) {
                    v += bias[n];
                } else {
                    float s = bng[n] * rsqrtf(bnv[n] + 1e-5f);
                    v = v * s + (cbias[n] - bnm[n]) * s + bnb[n];
                    v = fmaxf(v, 0.f);
                }
                C[(long)m * N + n] = v;
            }
        }
    }
}

// ---------------------------------------------------------------------------
// Fused conv stems + BN + ReLU + half-pixel bilinear resize to [B,25,156,24].
// Channels 0-7: conv1a(5x2) on mfcc_t; 8-15: conv1b(2x6) on mfcc_f;
// 16-23: conv1b reused on mfcc_c (faithful to reference).
// ---------------------------------------------------------------------------
__device__ __forceinline__ float stem_conv(const float* in, int b, int y, int x,
                                           const float* w, int KH, int KW,
                                           float cb, float sc, float sh)
{
    float s = 0.f;
    for (int ky = 0; ky < KH; ++ky)
        for (int kx = 0; kx < KW; ++kx)
            s += in[(long)b * 8000 + (y + ky) * 40 + (x + kx)] * w[ky * KW + kx];
    s = (s + cb) * sc + sh;
    return fmaxf(s, 0.f);
}

__global__ void stems_resize(const float* mt, const float* mf, const float* mc,
    const float* w1a, const float* b1a, const float* g1a, const float* be1a,
    const float* m1a, const float* v1a,
    const float* w1b, const float* b1b, const float* g1b, const float* be1b,
    const float* m1b, const float* v1b,
    float* xcat)
{
    long id = (long)blockIdx.x * blockDim.x + threadIdx.x;
    const long total = 128L * 25 * 156 * 24;
    if (id >= total) return;
    int c = (int)(id % 24); long r = id / 24;
    int ox = (int)(r % 156); r /= 156;
    int oy = (int)(r % 25);  int b = (int)(r / 25);
    int s = c >> 3, co = c & 7;
    const float* in = (s == 0) ? mt : ((s == 1) ? mf : mc);
    const float* w; float cb, sc, sh; int KH, KW, Hs, Ws;
    if (s == 0) {
        w = w1a + co * 10; cb = b1a[co];
        sc = g1a[co] * rsqrtf(v1a[co] + 1e-5f); sh = be1a[co] - m1a[co] * sc;
        KH = 5; KW = 2; Hs = 196; Ws = 39;
    } else {
        w = w1b + co * 12; cb = b1b[co];
        sc = g1b[co] * rsqrtf(v1b[co] + 1e-5f); sh = be1b[co] - m1b[co] * sc;
        KH = 2; KW = 6; Hs = 199; Ws = 35;
    }
    float fy = (oy + 0.5f) * ((float)Hs / 25.f)  - 0.5f;
    float fx = (ox + 0.5f) * ((float)Ws / 156.f) - 0.5f;
    float y0f = floorf(fy), x0f = floorf(fx);
    float ty = fy - y0f, tx = fx - x0f;
    int y0 = iclampi((int)y0f,     0, Hs - 1);
    int y1 = iclampi((int)y0f + 1, 0, Hs - 1);
    int x0 = iclampi((int)x0f,     0, Ws - 1);
    int x1 = iclampi((int)x0f + 1, 0, Ws - 1);
    float v00 = stem_conv(in, b, y0, x0, w, KH, KW, cb, sc, sh);
    float v01 = stem_conv(in, b, y0, x1, w, KH, KW, cb, sc, sh);
    float v10 = stem_conv(in, b, y1, x0, w, KH, KW, cb, sc, sh);
    float v11 = stem_conv(in, b, y1, x1, w, KH, KW, cb, sc, sh);
    xcat[id] = (1.f - ty) * ((1.f - tx) * v00 + tx * v01)
             +        ty  * ((1.f - tx) * v10 + tx * v11);
}

// 2x2 max pool on NHWC (VALID: floor division)
__global__ void maxpool2x2(const float* in, float* out, int Bz, int H, int W, int C)
{
    long id = (long)blockIdx.x * blockDim.x + threadIdx.x;
    int Ho = H / 2, Wo = W / 2;
    long total = (long)Bz * Ho * Wo * C;
    if (id >= total) return;
    int c = (int)(id % C); long r = id / C;
    int x = (int)(r % Wo); r /= Wo;
    int y = (int)(r % Ho); int b = (int)(r / Ho);
    const float* p = in + (((long)b * H + 2 * y) * W + 2 * x) * C + c;
    long rs = (long)W * C;
    out[id] = fmaxf(fmaxf(p[0], p[C]), fmaxf(p[rs], p[rs + C]));
}

// Per-position channel softmax(Q*K)*V, reduced (mean/sum) over R rows per b.
__global__ void attn_reduce(const float* __restrict__ Q, const float* __restrict__ K,
                            const float* __restrict__ V, float* outbase,
                            int R, float scale, int ostride)
{
    __shared__ float red[128];
    int b = blockIdx.x, t = threadIdx.x;
    float acc = 0.f;
    for (int rr = 0; rr < R; ++rr) {
        long row = (long)(b * R + rr) * 96;
        float q = 0.f, k = 0.f, v = 0.f, val = -1e30f;
        if (t < 96) { q = Q[row + t]; k = K[row + t]; v = V[row + t]; val = q * k; }
        red[t] = val; __syncthreads();
        for (int s = 64; s > 0; s >>= 1) {
            if (t < s) red[t] = fmaxf(red[t], red[t + s]);
            __syncthreads();
        }
        float mx = red[0]; __syncthreads();
        float e = (t < 96) ? __expf(val - mx) : 0.f;
        red[t] = e; __syncthreads();
        for (int s = 64; s > 0; s >>= 1) {
            if (t < s) red[t] += red[t + s];
            __syncthreads();
        }
        float sum = red[0]; __syncthreads();
        if (t < 96) acc += (e / sum) * v;
    }
    if (t < 96) outbase[(long)b * ostride + t] = acc * scale;
}

// Embedding gather: Ex[(tau*128 + b)*200 + e] = emb[tokens[b*50+tau]*200 + e]
__global__ void embed_gather(const int* tokens, const float* emb, float* Ex)
{
    long id = (long)blockIdx.x * blockDim.x + threadIdx.x;
    const long total = 50L * 128 * 200;
    if (id >= total) return;
    int e = (int)(id % 200); long r = id / 200;
    int b = (int)(r % 128); int tau = (int)(r / 128);
    Ex[id] = emb[(long)tokens[b * 50 + tau] * 200 + e];
}

__global__ void zero_f32(float* p, long n)
{
    long id = (long)blockIdx.x * blockDim.x + threadIdx.x;
    if (id < n) p[id] = 0.f;
}

// One GRU timestep (torch gate order r,z,n; b_ih already folded into gi).
__global__ void gru_step(const float* __restrict__ gi, const float* __restrict__ hprev,
                         const float* __restrict__ whh, const float* __restrict__ bhh,
                         float* hnew, float* outs, int tau)
{
    int id = blockIdx.x * blockDim.x + threadIdx.x;
    if (id >= 128 * 100) return;
    int b = id / 100, j = id % 100;
    const float* hp = hprev + b * 100;
    float ar = bhh[j], az = bhh[100 + j], an = bhh[200 + j];
    for (int i = 0; i < 100; ++i) {
        float hv = hp[i];
        ar += hv * whh[j * 100 + i];
        az += hv * whh[(100 + j) * 100 + i];
        an += hv * whh[(200 + j) * 100 + i];
    }
    const float* gr = gi + ((long)tau * 128 + b) * 300;
    float rg = 1.f / (1.f + __expf(-(gr[j] + ar)));
    float zg = 1.f / (1.f + __expf(-(gr[100 + j] + az)));
    float ng = tanhf(gr[200 + j] + rg * an);
    float h2 = (1.f - zg) * ng + zg * hp[j];
    hnew[id] = h2;
    outs[((long)b * 50 + tau) * 100 + j] = h2;
}

// x_gap[b,c] = mean over 234 positions of x4 [p,96]
__global__ void gap96(const float* x4, float* xgap)
{
    int b = blockIdx.x, c = threadIdx.x;  // blockDim = 96
    float a = 0.f;
    for (int s = 0; s < 234; ++s) a += x4[((long)b * 234 + s) * 96 + c];
    xgap[b * 96 + c] = a * (1.f / 234.f);
}

// feat[b, 3168+t] = softmax(x_gap @ w_fa^T + b_fa)[t] * h[b,t]   (t<100)
__global__ void gate_text(const float* xgap, const float* wfa, const float* bfa,
                          const float* h, float* feat)
{
    __shared__ float red[128];
    int b = blockIdx.x, t = threadIdx.x;
    float val = -1e30f;
    if (t < 100) {
        float s = bfa[t];
        for (int i = 0; i < 96; ++i) s += xgap[b * 96 + i] * wfa[t * 96 + i];
        val = s;
    }
    red[t] = val; __syncthreads();
    for (int s = 64; s > 0; s >>= 1) { if (t < s) red[t] = fmaxf(red[t], red[t + s]); __syncthreads(); }
    float mx = red[0]; __syncthreads();
    float e = (t < 100) ? __expf(val - mx) : 0.f;
    red[t] = e; __syncthreads();
    for (int s = 64; s > 0; s >>= 1) { if (t < s) red[t] += red[t + s]; __syncthreads(); }
    float sum = red[0];
    if (t < 100) feat[(long)b * 3268 + 3168 + t] = (e / sum) * h[b * 100 + t];
}

// feat[b, 3072+t] = softmax(h @ w_ft^T + b_ft)[t] * x_gap[b,t]   (t<96)
__global__ void gate_audio(const float* h, const float* wft, const float* bft,
                           const float* xgap, float* feat)
{
    __shared__ float red[128];
    int b = blockIdx.x, t = threadIdx.x;
    float val = -1e30f;
    if (t < 96) {
        float s = bft[t];
        for (int i = 0; i < 100; ++i) s += h[b * 100 + i] * wft[t * 100 + i];
        val = s;
    }
    red[t] = val; __syncthreads();
    for (int s = 64; s > 0; s >>= 1) { if (t < s) red[t] = fmaxf(red[t], red[t + s]); __syncthreads(); }
    float mx = red[0]; __syncthreads();
    float e = (t < 96) ? __expf(val - mx) : 0.f;
    red[t] = e; __syncthreads();
    for (int s = 64; s > 0; s >>= 1) { if (t < s) red[t] += red[t + s]; __syncthreads(); }
    float sum = red[0];
    if (t < 96) feat[(long)b * 3268 + 3072 + t] = (e / sum) * xgap[b * 96 + t];
}

// logits[b,e] = feat[b] . w_e[e] + b_e[e]
__global__ void final_fc(const float* feat, const float* we, const float* bev, float* out)
{
    int id = blockIdx.x * blockDim.x + threadIdx.x;
    if (id >= 512) return;
    int b = id / 4, e = id % 4;
    float s = bev[e];
    for (int i = 0; i < 3268; ++i) s += feat[(long)b * 3268 + i] * we[e * 3268 + i];
    out[id] = s;
}

// ---------------------------------------------------------------------------
extern "C" void kernel_launch(void* const* d_in, const int* in_sizes, int n_in,
                              void* d_out, int out_size, void* d_ws, size_t ws_size,
                              hipStream_t stream)
{
    (void)in_sizes; (void)n_in; (void)out_size; (void)ws_size;
    auto f = [&](int i) { return (const float*)d_in[i]; };
    const float* mfcc_t = f(0);
    const float* mfcc_f = f(1);
    const float* mfcc_c = f(2);
    const int*   tokens = (const int*)d_in[3];
    const float *w1a = f(5),  *b1a = f(6),  *g1a = f(7),  *be1a = f(8),  *m1a = f(9),  *v1a = f(10);
    const float *w1b = f(11), *b1b = f(12), *g1b = f(13), *be1b = f(14), *m1b = f(15), *v1b = f(16);
    const float *w2 = f(17), *b2 = f(18), *g2 = f(19), *be2 = f(20), *m2 = f(21), *v2 = f(22);
    const float *w3 = f(23), *b3 = f(24), *g3 = f(25), *be3 = f(26), *m3 = f(27), *v3 = f(28);
    const float *w4 = f(29), *b4 = f(30), *g4 = f(31), *be4 = f(32), *m4 = f(33), *v4 = f(34);
    const float* emb  = f(47);
    const float *w_ih = f(48), *w_hh = f(49), *b_ih = f(50), *b_hh = f(51);
    const float *w_fa = f(64), *b_fa = f(65), *w_ft = f(66), *b_ft = f(67);
    const float *w_e = f(68), *b_e = f(69);

    float* ws = (float*)d_ws;
    // Region R1 (stems output, later reused by the text branch scratch):
    float* xcat = ws;                          // 11,980,800
    float* Ex   = xcat;                        //  1,280,000 (after conv2 consumed xcat)
    float* gi   = xcat + 1280000;              //  1,920,000
    float* outs = xcat + 3200000;              //    640,000
    float* tq   = xcat + 3840000;              //    614,400
    float* tk   = xcat + 4454400;              //    614,400
    float* tv   = xcat + 5068800;              //    614,400
    // Region R0 (conv2 out -> conv3 out -> audio Q/K/V):
    float* R0 = ws + 11980800;                 // 15,974,400
    float* y2 = R0;
    float* y3 = R0;
    float* Qb = R0;
    float* Kb = R0 + 2875392;
    float* Vb = R0 + 2 * 2875392;
    // Persistent:
    float* p2   = ws + 11980800 + 15974400;    //  3,833,856
    float* p3   = p2 + 3833856;                //  1,916,928
    float* x4   = p3 + 1916928;                //  2,875,392
    float* feat = x4 + 2875392;                //    418,304
    float* xgap = feat + 418304;               //     12,288
    float* hA   = xgap + 12288;                //     12,800
    float* hB   = hA + 12800;                  //     12,800
    float* w2r  = hB + 12800;                  //      6,912  (32*216)
    float* w3r  = w2r + 6912;                  //     18,432  (64*288)
    float* w4r  = w3r + 18432;                 //     55,296  (96*576)

    auto gemm = [&](const float* A, const float* Bt, float* C, int M, int N, int K,
                    int im2col, int H, int W, int Cin, int epi,
                    const float* bias, const float* bng, const float* bnb,
                    const float* bnm, const float* bnv, const float* cbias) {
        int tiles = ((M + 15) / 16) * ((N + 31) / 32);
        dim3 grid((tiles + 3) / 4), blk(32, 4);
        gemm16_nt<<<grid, blk, 0, stream>>>(A, Bt, C, M, N, K, im2col, H, W, Cin,
                                            epi, bias, bng, bnb, bnm, bnv, cbias);
    };

    // ---- weight repack for implicit-GEMM convs ----
    repack_w9<<<(6912 + 255) / 256, 256, 0, stream>>>(w2, w2r, 32, 24);
    repack_w9<<<(18432 + 255) / 256, 256, 0, stream>>>(w3, w3r, 64, 32);
    repack_w9<<<(55296 + 255) / 256, 256, 0, stream>>>(w4, w4r, 96, 64);

    // ---- audio CNN ----
    {
        long total = 128L * 25 * 156 * 24;
        stems_resize<<<(int)((total + 255) / 256), 256, 0, stream>>>(
            mfcc_t, mfcc_f, mfcc_c,
            w1a, b1a, g1a, be1a, m1a, v1a,
            w1b, b1b, g1b, be1b, m1b, v1b, xcat);
    }
    gemm(xcat, w2r, y2, 128 * 25 * 156, 32, 24 * 9, 1, 25, 156, 24, 1,
         nullptr, g2, be2, m2, v2, b2);
    {
        long total = 128L * 12 * 78 * 32;
        maxpool2x2<<<(int)((total + 255) / 256), 256, 0, stream>>>(y2, p2, 128, 25, 156, 32);
    }
    gemm(p2, w3r, y3, 128 * 12 * 78, 64, 32 * 9, 1, 12, 78, 32, 1,
         nullptr, g3, be3, m3, v3, b3);
    {
        long total = 128L * 6 * 39 * 64;
        maxpool2x2<<<(int)((total + 255) / 256), 256, 0, stream>>>(y3, p3, 128, 12, 78, 64);
    }
    gemm(p3, w4r, x4, 128 * 234, 96, 64 * 9, 1, 6, 39, 64, 1,
         nullptr, g4, be4, m4, v4, b4);
    gap96<<<128, 96, 0, stream>>>(x4, xgap);

    // ---- text branch: embedding + GRU ----
    embed_gather<<<(1280000 + 255) / 256, 256, 0, stream>>>(tokens, emb, Ex);
    gemm(Ex, w_ih, gi, 6400, 300, 200, 0, 0, 0, 0, 0, b_ih,
         nullptr, nullptr, nullptr, nullptr, nullptr);
    zero_f32<<<50, 256, 0, stream>>>(hA, 12800);
    float* cur = hA; float* nxt = hB;
    for (int tau = 0; tau < 50; ++tau) {
        gru_step<<<50, 256, 0, stream>>>(gi, cur, w_hh, b_hh, nxt, outs, tau);
        float* tmp = cur; cur = nxt; nxt = tmp;
    }
    float* hfin = cur;

    // ---- audio attention (2 stages x 8 heads) ----
    for (int s = 0; s < 2; ++s) {
        const float* wq = f(35 + s * 6); const float* bq = f(36 + s * 6);
        const float* wk = f(37 + s * 6); const float* bk = f(38 + s * 6);
        const float* wv = f(39 + s * 6); const float* bv = f(40 + s * 6);
        for (int h = 0; h < 8; ++h) {
            gemm(x4, wq + h * 9216, Qb, 29952, 96, 96, 0, 0, 0, 0, 0, bq + h * 96,
                 nullptr, nullptr, nullptr, nullptr, nullptr);
            gemm(x4, wk + h * 9216, Kb, 29952, 96, 96, 0, 0, 0, 0, 0, bk + h * 96,
                 nullptr, nullptr, nullptr, nullptr, nullptr);
            gemm(x4, wv + h * 9216, Vb, 29952, 96, 96, 0, 0, 0, 0, 0, bv + h * 96,
                 nullptr, nullptr, nullptr, nullptr, nullptr);
            attn_reduce<<<128, 128, 0, stream>>>(Qb, Kb, Vb,
                feat + h * 192 + s * 96, 234, 1.f / 234.f, 3268);
        }
    }

    // ---- text attention (2 stages x 8 heads) ----
    for (int s = 0; s < 2; ++s) {
        const float* wq = f(52 + s * 6); const float* bq = f(53 + s * 6);
        const float* wk = f(54 + s * 6); const float* bk = f(55 + s * 6);
        const float* wv = f(56 + s * 6); const float* bv = f(57 + s * 6);
        for (int h = 0; h < 8; ++h) {
            gemm(outs, wq + h * 9600, tq, 6400, 96, 100, 0, 0, 0, 0, 0, bq + h * 96,
                 nullptr, nullptr, nullptr, nullptr, nullptr);
            gemm(outs, wk + h * 9600, tk, 6400, 96, 100, 0, 0, 0, 0, 0, bk + h * 96,
                 nullptr, nullptr, nullptr, nullptr, nullptr);
            gemm(outs, wv + h * 9600, tv, 6400, 96, 100, 0, 0, 0, 0, 0, bv + h * 96,
                 nullptr, nullptr, nullptr, nullptr, nullptr);
            attn_reduce<<<128, 128, 0, stream>>>(tq, tk, tv,
                feat + 1536 + s * 768 + h * 96, 50, 1.f, 3268);
        }
    }

    // ---- gated cross-modal fusion + classifier ----
    gate_audio<<<128, 128, 0, stream>>>(hfin, w_ft, b_ft, xgap, feat);
    gate_text<<<128, 128, 0, stream>>>(xgap, w_fa, b_fa, hfin, feat);
    final_fc<<<2, 256, 0, stream>>>(feat, w_e, b_e, (float*)d_out);
}
